// OCOR_66649302499964
// MI455X (gfx1250) — matrix operand order repaired and checked by reading, not compiled, not run on hardware
//
// MI455X (gfx1250) — cross-batch pixel attention block, bf16 WMMA implicit-GEMM.
// Compute-bound (~1.24 TFLOP vs ~0.5 GB traffic @ 23.3 TB/s) -> all matmuls via
// v_wmma_f32_16x16x32_bf16 with f32 accumulation. Conv wave-tile is 4 pixels x
// 64 cols (16 accumulators) -> 1.0 b128 load per WMMA to stay under L0 bandwidth.
#include <hip/hip_runtime.h>
#include <hip/hip_bf16.h>

typedef __bf16 bf16_t;
typedef __attribute__((ext_vector_type(16))) __bf16 v16bf;
typedef __attribute__((ext_vector_type(8)))  __bf16 v8bf;
typedef __attribute__((ext_vector_type(8)))  float  v8f;

#define HID  512
#define NB   16
#define HW   1024
#define CH   2048   // q/k/v channels (4*HID)
#define CQKV 6144   // fused q|k|v columns

__device__ __forceinline__ v16bf frag_zero() {
  v16bf z;
#pragma unroll
  for (int i = 0; i < 16; ++i) z[i] = (bf16_t)0.0f;
  return z;
}

// A-matrix 16x32 bf16 fragment: lane group g (=lane/16) holds K = g*16..g*16+15
// in elements 0..15 (contiguous order, matching the B fragments).
__device__ __forceinline__ v16bf load_a_frag(const bf16_t* __restrict__ rowk, int g) {
  v8bf lo = *(const v8bf*)(rowk + g * 16);
  v8bf hi = *(const v8bf*)(rowk + g * 16 + 8);
  v16bf r;
#pragma unroll
  for (int i = 0; i < 8; ++i) { r[i] = lo[i]; r[i + 8] = hi[i]; }
  return r;
}

// ---------------- layout transform kernels (fp32 -> bf16) ----------------

// x (B,C,H,W) fp32 -> xT [p][n][c] bf16 (c contiguous)
__global__ void xT_kernel(const float* __restrict__ x, bf16_t* __restrict__ xT, int n) {
  int i = blockIdx.x * blockDim.x + threadIdx.x;
  if (i >= n) return;
  int c = i & (HID - 1);
  int b = (i >> 9) & 15;
  int p = i >> 13;
  xT[i] = (bf16_t)x[((size_t)b * HID + c) * HW + p];
}

// Wq|Wk|Wv (2048,512,3,3) -> Wt [co 0..6143][kpos*512+ci] bf16
__global__ void wqkv_kernel(const float* __restrict__ Wq, const float* __restrict__ Wk,
                            const float* __restrict__ Wv, bf16_t* __restrict__ out, int n) {
  int i = blockIdx.x * blockDim.x + threadIdx.x;
  if (i >= n) return;
  int k  = i % 4608;        // 9*512
  int co = i / 4608;
  int kpos = k >> 9, ci = k & 511;
  const float* W = Wq; int c = co;
  if (co >= 4096)      { W = Wv; c = co - 4096; }
  else if (co >= 2048) { W = Wk; c = co - 2048; }
  out[i] = (bf16_t)W[((size_t)c * HID + ci) * 9 + kpos];
}

// Wo (512,2048,3,3) -> Wot [co][kpos*2048+ci] bf16
__global__ void wo_kernel(const float* __restrict__ Wo, bf16_t* __restrict__ out, int n) {
  int i = blockIdx.x * blockDim.x + threadIdx.x;
  if (i >= n) return;
  int k  = i % 18432;       // 9*2048
  int co = i / 18432;
  int kpos = k >> 11, ci = k & 2047;
  out[i] = (bf16_t)Wo[((size_t)co * CH + ci) * 9 + kpos];
}

// ---------------- implicit-GEMM 3x3 SAME conv via WMMA ----------------
// A: [HW][NB][CIN] bf16 (activations), Wt: [COUT][9*CIN] bf16.
// M-tile = one pixel x 16 batches; wave tile = 4 pixels x 64 cols (16 WMMA accs,
// 16 b128 loads per 16 WMMAs). Block = 4 waves = 4 pixels x 256 cols. K-chunks
// of 32 never cross a kpos boundary (CIN % 32 == 0), so padding checks are
// per-kpos and wave-uniform (EXEC stays all-ones around every WMMA).
template <int CIN, bool RESID>
__global__ __launch_bounds__(128) void conv3x3_wmma(
    const bf16_t* __restrict__ A, const bf16_t* __restrict__ Wt,
    bf16_t* __restrict__ outB,                 // !RESID: [HW][NB][COUT] bf16
    float* __restrict__ outF,                  // RESID : d_out (B,COUT,H,W) f32
    const float* __restrict__ xres, int COUT)
{
  const int lane = threadIdx.x & 31;
  const int wv   = threadIdx.x >> 5;
  const int g    = lane >> 4;
  const int lr   = lane & 15;
  const int K9   = 9 * CIN;
  const int p0   = blockIdx.x * 4;
  const int col0 = blockIdx.y * 256 + wv * 64;

  v8f acc[4][4];
#pragma unroll
  for (int i = 0; i < 4; ++i)
#pragma unroll
    for (int j = 0; j < 4; ++j) acc[i][j] = {};

  // B fragment row pointers: per-lane column co = col0+t*16+lr, K chunk at g*16.
  const bf16_t* wrow[4];
#pragma unroll
  for (int t = 0; t < 4; ++t)
    wrow[t] = Wt + (size_t)(col0 + t * 16 + lr) * K9 + g * 16;

#pragma unroll
  for (int kpos = 0; kpos < 9; ++kpos) {
    const int kh = kpos / 3 - 1, kw = kpos % 3 - 1;
    const bf16_t* arow[4];
#pragma unroll
    for (int mp = 0; mp < 4; ++mp) {
      int p = p0 + mp;
      int h = (p >> 5) + kh, w = (p & 31) + kw;
      bool ok = ((unsigned)h < 32u) && ((unsigned)w < 32u);
      arow[mp] = ok ? (A + (size_t)((h * 32 + w) * NB + lr) * CIN) : nullptr;
    }
    const int kbase = kpos * CIN;
    for (int c0 = 0; c0 < CIN; c0 += 32) {
      v16bf a[4];
#pragma unroll
      for (int mp = 0; mp < 4; ++mp)
        a[mp] = arow[mp] ? load_a_frag(arow[mp] + c0, g) : frag_zero();
#pragma unroll
      for (int t = 0; t < 4; ++t) {
        v16bf b = *(const v16bf*)(wrow[t] + kbase + c0);
#pragma unroll
        for (int mp = 0; mp < 4; ++mp)
          acc[mp][t] = __builtin_amdgcn_wmma_f32_16x16x32_bf16(
              false, a[mp], false, b, (short)0, acc[mp][t], false, false);
      }
    }
  }

  // C layout: VGPR r, lanes 0-15 -> (M=r, N=lane); lanes 16-31 -> (M=r+8, N=lane-16)
#pragma unroll
  for (int mp = 0; mp < 4; ++mp) {
    int p = p0 + mp;
#pragma unroll
    for (int t = 0; t < 4; ++t) {
      int col = col0 + t * 16 + lr;
#pragma unroll
      for (int r = 0; r < 8; ++r) {
        int m = r + 8 * g;     // batch index
        float v = acc[mp][t][r];
        if (RESID) {
          size_t idx = ((size_t)m * COUT + col) * HW + p;
          outF[idx] = xres[idx] + v;
        } else {
          outB[(size_t)(p * NB + m) * COUT + col] = (bf16_t)v;
        }
      }
    }
  }
}

// ---------------- per-pixel cross-batch attention (1 wave / pixel) -----------
// att(16x16) = softmax_j( Q(16x2048) K^T / sqrt(512) ); virt = att V(16x2048).
// Also accumulates per-batch sum/sumsq of virt for GroupNorm.
__global__ __launch_bounds__(256) void attention_kernel(
    const bf16_t* __restrict__ qkv,   // [HW][NB][6144]
    bf16_t* __restrict__ vraw,        // [HW][NB][2048]
    float* __restrict__ stats)        // [NB][2] (sum, sumsq)
{
  __shared__ __align__(64) bf16_t att_s[8][256];
  __shared__ __align__(64) bf16_t vt_s[8][256];
  const int lane = threadIdx.x & 31;
  const int wv   = threadIdx.x >> 5;
  const int g    = lane >> 4;
  const int lr   = lane & 15;
  const int p    = blockIdx.x * 8 + wv;

  const bf16_t* row = qkv + (size_t)(p * NB + lr) * CQKV;  // batch lr: q row (A) & k row (B col)

  // --- Q.K^T, K = 2048 ---
  v8f att = {};
  for (int k0 = 0; k0 < CH; k0 += 32) {
    v16bf a = load_a_frag(row + k0, g);
    v16bf b = *(const v16bf*)(row + CH + k0 + g * 16);
    att = __builtin_amdgcn_wmma_f32_16x16x32_bf16(false, a, false, b, (short)0, att, false, false);
  }

  // --- softmax over j (cols = lanes within each 16-lane half) ---
  const float scale = 0.044194173824159216f;  // 1/sqrt(512)
#pragma unroll
  for (int r = 0; r < 8; ++r) {
    float v = att[r] * scale;
    float mx = v;
#pragma unroll
    for (int mask = 1; mask < 16; mask <<= 1) mx = fmaxf(mx, __shfl_xor(mx, mask, 32));
    float e = __expf(v - mx);
    float s = e;
#pragma unroll
    for (int mask = 1; mask < 16; mask <<= 1) s += __shfl_xor(s, mask, 32);
    att[r] = e / s;
  }

  // round-trip att through LDS to re-lay-out as an A fragment (K=16 pad to 32)
#pragma unroll
  for (int r = 0; r < 8; ++r)
    att_s[wv][(r + 8 * g) * 16 + lr] = (bf16_t)att[r];
  __syncthreads();

  v16bf a_att = frag_zero();
  {
    if (g == 0) a_att = *(const v16bf*)(&att_s[wv][lr * 16]);
  }

  const bf16_t* vrow = qkv + (size_t)(p * NB + lr) * CQKV + 2 * CH;  // v row j=lr
  float psum[8], psq[8];
#pragma unroll
  for (int r = 0; r < 8; ++r) { psum[r] = 0.f; psq[r] = 0.f; }

  for (int c0 = 0; c0 < CH; c0 += 16) {
    // stage V tile transposed: vt[c_local][j] so B frags are contiguous per lane
    v8bf vvld = *(const v8bf*)(vrow + c0 + g * 8);
#pragma unroll
    for (int e = 0; e < 8; ++e)
      vt_s[wv][(g * 8 + e) * 16 + lr] = vvld[e];
    __syncthreads();

    v16bf b = frag_zero();                 // lanes 16-31 carry K=16..31 -> zero pad
    if (g == 0) b = *(const v16bf*)(&vt_s[wv][lr * 16]);
    v8f o = {};
    o = __builtin_amdgcn_wmma_f32_16x16x32_bf16(false, a_att, false, b, (short)0, o, false, false);
    __syncthreads();

#pragma unroll
    for (int r = 0; r < 8; ++r) {
      float v = o[r];
      psum[r] += v; psq[r] += v * v;
      int m = r + 8 * g;
      vraw[(size_t)(p * NB + m) * CH + c0 + lr] = (bf16_t)v;
    }
  }

  // per-row reduce across the 16-lane half, then 1 atomic pair per (wave,row)
#pragma unroll
  for (int r = 0; r < 8; ++r) {
    float s = psum[r], q = psq[r];
#pragma unroll
    for (int mask = 1; mask < 16; mask <<= 1) {
      s += __shfl_xor(s, mask, 32);
      q += __shfl_xor(q, mask, 32);
    }
    if (lr == 0) {
      int m = r + 8 * g;
      atomicAdd(&stats[2 * m], s);
      atomicAdd(&stats[2 * m + 1], q);
    }
  }
}

// ---------------- GroupNorm(1,C) + affine + ReLU -> bf16 ----------------
__global__ void norm_kernel(const bf16_t* __restrict__ vraw, const float* __restrict__ stats,
                            const float* __restrict__ gamma, const float* __restrict__ beta,
                            bf16_t* __restrict__ vout, int n) {
  int i = blockIdx.x * blockDim.x + threadIdx.x;
  if (i >= n) return;
  int c = i & (CH - 1);
  int b = (i >> 11) & 15;
  const float inv_cnt = 1.0f / (float)(CH * HW);
  float mu  = stats[2 * b] * inv_cnt;
  float var = stats[2 * b + 1] * inv_cnt - mu * mu;
  float y = ((float)vraw[i] - mu) * rsqrtf(var + 1e-5f);
  y = y * gamma[c] + beta[c];
  vout[i] = (bf16_t)fmaxf(y, 0.0f);
}

// ---------------- host-side launch ----------------
extern "C" void kernel_launch(void* const* d_in, const int* in_sizes, int n_in,
                              void* d_out, int out_size, void* d_ws, size_t ws_size,
                              hipStream_t stream) {
  (void)in_sizes; (void)n_in; (void)out_size; (void)ws_size;
  const float* x     = (const float*)d_in[0];
  const float* Wq    = (const float*)d_in[1];
  const float* Wk    = (const float*)d_in[2];
  const float* Wv    = (const float*)d_in[3];
  const float* Wo    = (const float*)d_in[4];
  const float* gamma = (const float*)d_in[5];
  const float* beta  = (const float*)d_in[6];

  char* ws = (char*)d_ws;
  size_t off = 0;
  auto alloc = [&](size_t bytes) -> void* {
    void* ptr = ws + off;
    off = (off + bytes + 255) & ~(size_t)255;
    return ptr;
  };
  bf16_t* xT    = (bf16_t*)alloc((size_t)HW * NB * HID * 2);          //  16.8 MB
  bf16_t* Wqkvt = (bf16_t*)alloc((size_t)CQKV * 9 * HID * 2);         //  56.6 MB
  bf16_t* Wot   = (bf16_t*)alloc((size_t)HID * 9 * CH * 2);           //  18.9 MB
  bf16_t* qkv   = (bf16_t*)alloc((size_t)HW * NB * CQKV * 2);         // 201.3 MB
  bf16_t* vraw  = (bf16_t*)alloc((size_t)HW * NB * CH * 2);           //  67.1 MB
  bf16_t* vnorm = (bf16_t*)alloc((size_t)HW * NB * CH * 2);           //  67.1 MB
  float*  stats = (float*)alloc(2 * NB * sizeof(float));

  hipMemsetAsync(stats, 0, 2 * NB * sizeof(float), stream);

  {
    int n = HW * NB * HID;
    xT_kernel<<<(n + 255) / 256, 256, 0, stream>>>(x, xT, n);
  }
  {
    int n = CQKV * 9 * HID;
    wqkv_kernel<<<(n + 255) / 256, 256, 0, stream>>>(Wq, Wk, Wv, Wqkvt, n);
  }
  {
    int n = HID * 9 * CH;
    wo_kernel<<<(n + 255) / 256, 256, 0, stream>>>(Wo, Wot, n);
  }

  // fused q|k|v conv: M=16384, N=6144, K=4608
  {
    dim3 grid(HW / 4, CQKV / 256);
    conv3x3_wmma<HID, false><<<grid, 128, 0, stream>>>(xT, Wqkvt, qkv, nullptr, nullptr, CQKV);
  }

  attention_kernel<<<HW / 8, 256, 0, stream>>>(qkv, vraw, stats);

  {
    int n = HW * NB * CH;
    norm_kernel<<<(n + 255) / 256, 256, 0, stream>>>(vraw, stats, gamma, beta, vnorm, n);
  }

  // output conv + residual: M=16384, N=512, K=18432
  {
    dim3 grid(HW / 4, HID / 256);
    conv3x3_wmma<CH, true><<<grid, 128, 0, stream>>>(vnorm, Wot, nullptr, (float*)d_out, x, HID);
  }
}
